// SimpleLSTM_64957085385395
// MI455X (gfx1250) — compile-verified, matrix operand order checked by reading, not run on hardware
//
#include <hip/hip_runtime.h>

// ---------------------------------------------------------------------------
// LSTM (H=64, input_dim=1) scan + final FC for gfx1250 (CDNA5, wave32, WMMA)
// Branch-free native transcendentals (v_tanh_f32) on the scan critical path.
// ---------------------------------------------------------------------------

typedef __attribute__((ext_vector_type(16))) _Float16 v16h;
typedef __attribute__((ext_vector_type(8)))  _Float16 v8h;
typedef __attribute__((ext_vector_type(8)))  float    v8f;

#define HDIM     64
#define TSTEPS   1024
#define BROWS    16          // batch rows per workgroup (= WMMA M)
#define NTHREADS 128         // 4 wave32 waves
#define XCH      256         // x staging chunk (time steps)
#define XSTRIDE  (XCH + 4)   // pad: banks spread (row*4+t) % 64
#define HSTRIDE  (HDIM + 8)  // pad h rows: 72 halves = 144B -> distinct banks

// Branch-free tanh: native v_tanh_f32 if available, else exp2+rcp (all
// single-issue TRANS/VALU ops, no EXEC-mask branching).
static __device__ __forceinline__ float fast_tanh(float x) {
#if __has_builtin(__builtin_amdgcn_tanhf)
    return __builtin_amdgcn_tanhf(x);
#else
    // tanh(x) = 2/(1+exp(-2x)) - 1 ; exp(-2x) = exp2(-2*log2(e)*x)
    const float e = __builtin_amdgcn_exp2f(x * -2.885390082f);
    return fmaf(2.0f, __builtin_amdgcn_rcpf(1.0f + e), -1.0f);
#endif
}

static __device__ __forceinline__ float fast_sigmoid(float x) {
    // sigmoid(x) = 0.5*tanh(0.5x) + 0.5
    return fmaf(0.5f, fast_tanh(0.5f * x), 0.5f);
}

__launch_bounds__(NTHREADS, 1)
__global__ void lstm_wmma_kernel(const float* __restrict__ x,
                                 const float* __restrict__ W_ih,
                                 const float* __restrict__ W_hh,
                                 const float* __restrict__ b_ih,
                                 const float* __restrict__ b_hh,
                                 const float* __restrict__ fc_W,
                                 const float* __restrict__ fc_b,
                                 float* __restrict__ out)
{
    __shared__ float    xs[BROWS * XSTRIDE];
    __shared__ _Float16 hbuf[2][BROWS][HSTRIDE];

    const int tid   = threadIdx.x;
    const int lane  = tid & 31;
    const int wave  = tid >> 5;          // 0..3: hidden cols [16w,16w+16) of each gate
    const int col16 = lane & 15;         // N within 16x16 tile / A-row for this lane
    const int half  = lane >> 4;         // lane group (K-half selector)
    const int row0  = blockIdx.x * BROWS;

    // ---- zero h(0) ----
    for (int i = tid; i < 2 * BROWS * HSTRIDE; i += NTHREADS)
        ((_Float16*)hbuf)[i] = (_Float16)0.0f;

    // ---- per-lane constants: this lane's gate column hcol of each gate ----
    const int hcol = 16 * wave + col16;          // hidden index within a gate
    float wih_c[4], bias_c[4];
#pragma unroll
    for (int g = 0; g < 4; ++g) {
        const int col = g * HDIM + hcol;         // column of the 4H gate vector
        wih_c[g]  = W_ih[col];
        bias_c[g] = b_ih[col] + b_hh[col];
    }

    // ---- B fragments (Whh_T), resident in VGPRs for all 1024 steps ----
    // Whh_T[k][col] = W_hh[col*H + k].
    // f16 B layout (mirrors documented 16-bit A layout): lane holds column
    // N = lane%16, half = lane/16; element e -> K = e + 8*half + (e>=8 ? 8:0),
    // +32 for the second K-chunk.
    v16h bfrag[4][2];
#pragma unroll
    for (int g = 0; g < 4; ++g) {
        const float* wcol = W_hh + (size_t)(g * HDIM + hcol) * HDIM;
#pragma unroll
        for (int ch = 0; ch < 2; ++ch) {
#pragma unroll
            for (int e = 0; e < 16; ++e) {
                const int k = e + half * 8 + ((e >= 8) ? 8 : 0) + 32 * ch;
                bfrag[g][ch][e] = (_Float16)wcol[k];
            }
        }
    }

    // ---- cell state lives in registers the whole scan ----
    // element r -> (batch row = r + 8*half, hidden col = hcol)
    float creg[8];
#pragma unroll
    for (int r = 0; r < 8; ++r) creg[r] = 0.0f;

    int cur = 0;
    for (int t0 = 0; t0 < TSTEPS; t0 += XCH) {
        // stage x chunk: coalesced float4 along time
        for (int i = tid; i < BROWS * (XCH / 4); i += NTHREADS) {
            const int r  = i / (XCH / 4);
            const int c4 = i % (XCH / 4);
            const float4 v =
                ((const float4*)(x + (size_t)(row0 + r) * TSTEPS + t0))[c4];
            float* dst = &xs[r * XSTRIDE + c4 * 4];
            dst[0] = v.x; dst[1] = v.y; dst[2] = v.z; dst[3] = v.w;
        }
        __syncthreads();

        for (int tl = 0; tl < XCH; ++tl) {
            // ---- A fragments: h(t) from LDS, documented 16-bit A layout ----
            const _Float16* hrow = &hbuf[cur][col16][0];
            const v8h p0 = *(const v8h*)(hrow +  0 + half * 8);
            const v8h p1 = *(const v8h*)(hrow + 16 + half * 8);
            const v8h p2 = *(const v8h*)(hrow + 32 + half * 8);
            const v8h p3 = *(const v8h*)(hrow + 48 + half * 8);
            const v16h a_lo = __builtin_shufflevector(
                p0, p1, 0,1,2,3,4,5,6,7,8,9,10,11,12,13,14,15);
            const v16h a_hi = __builtin_shufflevector(
                p2, p3, 0,1,2,3,4,5,6,7,8,9,10,11,12,13,14,15);

            // x for the 8 batch rows this lane's C/D elements map to
            float xv[8];
#pragma unroll
            for (int r = 0; r < 8; ++r)
                xv[r] = xs[(r + 8 * half) * XSTRIDE + tl];

            // ---- gates = (x*W_ih + b) + h @ Whh_T  via two chained WMMAs ----
            v8f acc[4];
#pragma unroll
            for (int g = 0; g < 4; ++g) {
                v8f c0;
#pragma unroll
                for (int r = 0; r < 8; ++r)
                    c0[r] = fmaf(xv[r], wih_c[g], bias_c[g]);
                v8f tmp = __builtin_amdgcn_wmma_f32_16x16x32_f16(
                    false, a_lo, false, bfrag[g][0], (short)0, c0, false, false);
                acc[g] = __builtin_amdgcn_wmma_f32_16x16x32_f16(
                    false, a_hi, false, bfrag[g][1], (short)0, tmp, false, false);
            }

            // ---- register-local cell update; write h(t+1) to other buffer --
#pragma unroll
            for (int r = 0; r < 8; ++r) {
                const float ig = fast_sigmoid(acc[0][r]);
                const float fg = fast_sigmoid(acc[1][r]);
                const float gg = fast_tanh(acc[2][r]);
                const float og = fast_sigmoid(acc[3][r]);
                const float cn = fmaf(fg, creg[r], ig * gg);
                creg[r] = cn;
                hbuf[cur ^ 1][r + 8 * half][hcol] = (_Float16)(og * fast_tanh(cn));
            }
            cur ^= 1;
            __syncthreads();   // covers h RAW (next read) and WAR (next write)
        }
    }

    // ---- final FC: out[row, cls] = h_T[row] . fc_W[cls] + fc_b[cls] ----
    if (tid < 32) {
        const int r   = tid & 15;
        const int cls = tid >> 4;
        float s = fc_b[cls];
#pragma unroll
        for (int k = 0; k < HDIM; ++k)
            s += (float)hbuf[cur][r][k] * fc_W[cls * HDIM + k];
        out[(size_t)(row0 + r) * 2 + cls] = s;
    }
}

extern "C" void kernel_launch(void* const* d_in, const int* in_sizes, int n_in,
                              void* d_out, int out_size, void* d_ws, size_t ws_size,
                              hipStream_t stream)
{
    const float* x    = (const float*)d_in[0];
    const float* W_ih = (const float*)d_in[1];
    const float* W_hh = (const float*)d_in[2];
    const float* b_ih = (const float*)d_in[3];
    const float* b_hh = (const float*)d_in[4];
    const float* fc_W = (const float*)d_in[5];
    const float* fc_b = (const float*)d_in[6];
    float* out = (float*)d_out;

    const int B = in_sizes[0] / TSTEPS;       // 512
    const int nblocks = B / BROWS;            // 32

    lstm_wmma_kernel<<<nblocks, NTHREADS, 0, stream>>>(
        x, W_ih, W_hh, b_ih, b_hh, fc_W, fc_b, out);
}
